// RWKVBlock_37374805409853
// MI455X (gfx1250) — matrix operand order repaired and checked by reading, not compiled, not run on hardware
//
#include <hip/hip_runtime.h>

// ---------------------------------------------------------------------------
// RWKV-4 block for MI455X (gfx1250): bf16 WMMA GEMMs with async-to-LDS weight
// staging + fused WKV scan.  B=8, T=2048, D=1024, FF=4096, M=B*T=16384.
// ---------------------------------------------------------------------------

#define Bn  8
#define Tn  2048
#define Dn  1024
#define FFn 4096
#define Mn  (Bn * Tn)   // 16384 rows

#define KSTEP 32        // K per WMMA step
#define BM    128       // block M tile
#define BN    128       // block N tile (staged in LDS)

typedef __bf16 bf16;
typedef __attribute__((ext_vector_type(16))) __bf16 v16bf;
typedef __attribute__((ext_vector_type(8)))  __bf16 v8bf;
typedef __attribute__((ext_vector_type(8)))  float  v8f;

union AFrag { v16bf v; v8bf h[2]; };

// ---------------------------------------------------------------------------
// Weight convert + transpose: W[K,N] f32 -> Wt[N,K] bf16 (so B-matrix WMMA
// fragments are contiguous 32B runs per lane).
// ---------------------------------------------------------------------------
__global__ __launch_bounds__(256)
void wconv_kernel(const float* __restrict__ W, bf16* __restrict__ Wt, int K, int N) {
    size_t i = (size_t)blockIdx.x * blockDim.x + threadIdx.x;
    if (i >= (size_t)K * N) return;
    int n = (int)(i % N);
    int k = (int)(i / N);
    Wt[(size_t)n * K + k] = (bf16)W[i];
}

// ---------------------------------------------------------------------------
// LayerNorm: one block (256 threads) per row of D=1024.
// ---------------------------------------------------------------------------
__global__ __launch_bounds__(256)
void ln_kernel(const float* __restrict__ x, const float* __restrict__ g,
               const float* __restrict__ b, float* __restrict__ h) {
    const int row = blockIdx.x;
    const float* xr = x + (size_t)row * Dn;
    float* hr = h + (size_t)row * Dn;

    __shared__ float s_sum[256];
    __shared__ float s_sq[256];

    float vals[Dn / 256];
    float s = 0.f, ss = 0.f;
#pragma unroll
    for (int i = 0; i < Dn / 256; ++i) {
        float v = xr[threadIdx.x + 256 * i];
        vals[i] = v;
        s += v;
        ss += v * v;
    }
    s_sum[threadIdx.x] = s;
    s_sq[threadIdx.x] = ss;
    __syncthreads();
#pragma unroll
    for (int off = 128; off > 0; off >>= 1) {
        if (threadIdx.x < off) {
            s_sum[threadIdx.x] += s_sum[threadIdx.x + off];
            s_sq[threadIdx.x]  += s_sq[threadIdx.x + off];
        }
        __syncthreads();
    }
    const float mu  = s_sum[0] * (1.0f / Dn);
    const float var = s_sq[0] * (1.0f / Dn) - mu * mu;
    const float rs  = rsqrtf(var + 1e-5f);
#pragma unroll
    for (int i = 0; i < Dn / 256; ++i) {
        int d = threadIdx.x + 256 * i;
        hr[d] = (vals[i] - mu) * rs * g[d] + b[d];
    }
}

// ---------------------------------------------------------------------------
// Time-mix: xk/xv/xr = lerp(h[t-1], h[t], mix) in bf16 (zeros at t==0).
// ---------------------------------------------------------------------------
__global__ __launch_bounds__(256)
void mix3_kernel(const float* __restrict__ h,
                 const float* __restrict__ mk, const float* __restrict__ mv,
                 const float* __restrict__ mr,
                 bf16* __restrict__ xk, bf16* __restrict__ xv, bf16* __restrict__ xr) {
    size_t idx = (size_t)blockIdx.x * blockDim.x + threadIdx.x;
    if (idx >= (size_t)Mn * Dn) return;
    int d = (int)(idx % Dn);
    size_t row = idx / Dn;
    int t = (int)(row % Tn);
    float hc = h[idx];
    float hp = (t == 0) ? 0.f : h[idx - Dn];
    float a = mk[d]; xk[idx] = (bf16)(hc * a + hp * (1.f - a));
    float b = mv[d]; xv[idx] = (bf16)(hc * b + hp * (1.f - b));
    float c = mr[d]; xr[idx] = (bf16)(hc * c + hp * (1.f - c));
}

// Channel-mix variant: two outputs.
__global__ __launch_bounds__(256)
void mix2_kernel(const float* __restrict__ h,
                 const float* __restrict__ mk, const float* __restrict__ mr,
                 bf16* __restrict__ xk, bf16* __restrict__ xr) {
    size_t idx = (size_t)blockIdx.x * blockDim.x + threadIdx.x;
    if (idx >= (size_t)Mn * Dn) return;
    int d = (int)(idx % Dn);
    size_t row = idx / Dn;
    int t = (int)(row % Tn);
    float hc = h[idx];
    float hp = (t == 0) ? 0.f : h[idx - Dn];
    float a = mk[d]; xk[idx] = (bf16)(hc * a + hp * (1.f - a));
    float c = mr[d]; xr[idx] = (bf16)(hc * c + hp * (1.f - c));
}

// ---------------------------------------------------------------------------
// WKV scan: one thread per (b,d) channel, sequential over T.
// Fuses r*wkv and bf16 conversion (output feeds the wo GEMM as A-operand).
// ---------------------------------------------------------------------------
__global__ __launch_bounds__(256)
void wkv_kernel(const float* __restrict__ k, const float* __restrict__ v,
                const float* __restrict__ r,
                const float* __restrict__ time_decay,
                const float* __restrict__ time_first,
                bf16* __restrict__ rwkv) {
    int gid = blockIdx.x * blockDim.x + threadIdx.x;   // < B*D
    int d = gid % Dn;
    int b = gid / Dn;
    const float w = -__expf(time_decay[d]);
    const float u = time_first[d];
    float aa = 0.f, bb = 0.f, pp = -1e38f;
    size_t base = (size_t)b * Tn * Dn + d;
    for (int t = 0; t < Tn; ++t) {
        size_t idx = base + (size_t)t * Dn;
        float kt = k[idx];
        float vt = v[idx];
        float ww = u + kt;
        float p  = fmaxf(pp, ww);
        float e1 = __expf(pp - p);
        float e2 = __expf(ww - p);
        float out = (e1 * aa + e2 * vt) / (e1 * bb + e2);
        float ww2 = pp + w;
        float p2  = fmaxf(ww2, kt);
        float e1b = __expf(ww2 - p2);
        float e2b = __expf(kt - p2);
        aa = e1b * aa + e2b * vt;
        bb = e1b * bb + e2b;
        pp = p2;
        rwkv[idx] = (bf16)(r[idx] * out);   // r already sigmoid-ed
    }
}

// ---------------------------------------------------------------------------
// Async copy of one 16B chunk global -> LDS (gfx1250 ASYNCcnt path).
// Generic LDS pointers carry the hardware LDS byte offset in the low 32 bits
// ---------------------------------------------------------------------------
__device__ __forceinline__ void async_g2l_b128(const bf16* g, const bf16* l) {
    unsigned lds_off = (unsigned)(uintptr_t)(const void*)l;
    asm volatile("global_load_async_to_lds_b128 %0, %1, off"
                 :: "v"(lds_off), "v"(g)
                 : "memory");
}

__device__ __forceinline__ void wait_asynccnt0() {
    asm volatile("s_wait_asynccnt 0x0" ::: "memory");
}

// ---------------------------------------------------------------------------
// WMMA GEMM: out[M,N] = A[M,K] (bf16) * Wt[N,K]^T (bf16), f32 accumulate.
// Block: 256 threads = 8 waves (4 along M x 2 along N), tile 128(M) x 128(N);
// each wave owns 32x64 (2x4 v_wmma_f32_16x16x32_bf16 accumulators).
// Weight tile (128 x 32 bf16 = 8KB) is double-buffered in LDS via
// global_load_async_to_lds_b128, overlapped with compute.  K-loop explicitly
// unrolled 2x (uniform trip count, so the in-loop barrier is safe) for a
// denser WMMA stream and less loop-control overhead.
// Epilogues: 0 = store f32; 1 = sigmoid f32; 2 = res + acc;
//            3 = res + gate*acc; 4 = relu(acc)^2 -> bf16.
// ---------------------------------------------------------------------------
template <int EPI>
__global__ __launch_bounds__(256)
void wmma_gemm_kernel(const bf16* __restrict__ A, const bf16* __restrict__ Bt,
                      float* __restrict__ outf, bf16* __restrict__ outb,
                      const float* __restrict__ res, const float* __restrict__ gate,
                      int N, int K) {
    __shared__ __attribute__((aligned(128))) bf16 smem[2][BN * KSTEP];

    const int tid  = threadIdx.x;
    const int wave = tid >> 5;
    const int lane = tid & 31;
    const int wm = wave & 3;          // 4 waves along M
    const int wn = wave >> 2;         // 2 waves along N (64 cols each)
    const int m0 = blockIdx.x * BM + wm * 32;
    const int nb = blockIdx.y * BN;   // block N base
    const int half = lane >> 4;       // 0: lanes 0-15, 1: lanes 16-31
    const int l16  = lane & 15;

    // Staging assignment: thread copies 32B of one weight row per K-step.
    const int srow = tid >> 1;            // 0..127 (local n)
    const int scol = (tid & 1) * 16;      // element offset 0 or 16
    const bf16* gsrc = Bt + (size_t)(nb + srow) * K + scol;

    v8f acc[2][4];
#pragma unroll
    for (int i = 0; i < 2; ++i)
#pragma unroll
        for (int j = 0; j < 4; ++j)
#pragma unroll
            for (int e = 0; e < 8; ++e) acc[i][j][e] = 0.f;

    // A rows for the two 16-row subtiles this wave owns.
    const bf16* arow0 = A + (size_t)(m0 + l16) * K;
    const bf16* arow1 = A + (size_t)(m0 + 16 + l16) * K;

    const int ksteps = K / KSTEP;

    // Prologue: stage first weight tile.
    {
        const bf16* l = &smem[0][srow * KSTEP + scol];
        async_g2l_b128(gsrc, l);
        async_g2l_b128(gsrc + 8, l + 8);
    }

#pragma unroll 2
    for (int kt = 0; kt < ksteps; ++kt) {
        const int k = kt * KSTEP;
        wait_asynccnt0();        // stage(kt) landed in LDS (this wave's copies)
        __syncthreads();         // all waves' copies visible; compute(kt-1) done

        if (kt + 1 < ksteps) {   // stage(kt+1) overlapped with compute(kt)
            const bf16* g = gsrc + (size_t)(kt + 1) * KSTEP;
            const bf16* l = &smem[(kt + 1) & 1][srow * KSTEP + scol];
            async_g2l_b128(g, l);
            async_g2l_b128(g + 8, l + 8);
        }

        // A fragment: lane<16 holds K k..k+15;
        //             lane>=16 holds K k+16..k+31.
        AFrag a0, a1;
        a0.h[0] = *(const v8bf*)(arow0 + k + half * 16);
        a0.h[1] = *(const v8bf*)(arow0 + k + half * 16 + 8);
        a1.h[0] = *(const v8bf*)(arow1 + k + half * 16);
        a1.h[1] = *(const v8bf*)(arow1 + k + half * 16 + 8);

        // B fragments from LDS: lane<16 -> K 0..15 of col n, lane>=16 -> K 16..31.
        const bf16* bbase = &smem[kt & 1][0];
#pragma unroll
        for (int j = 0; j < 4; ++j) {
            v16bf bj = *(const v16bf*)(bbase +
                (wn * 64 + j * 16 + l16) * KSTEP + half * 16);
            acc[0][j] = __builtin_amdgcn_wmma_f32_16x16x32_bf16(
                false, a0.v, false, bj, (short)0, acc[0][j], false, false);
            acc[1][j] = __builtin_amdgcn_wmma_f32_16x16x32_bf16(
                false, a1.v, false, bj, (short)0, acc[1][j], false, false);
        }
    }

    // Epilogue. C/D layout: VGPR r -> (M = mtile + r + 8*half, N = ntile + l16).
#pragma unroll
    for (int i = 0; i < 2; ++i) {
#pragma unroll
        for (int j = 0; j < 4; ++j) {
#pragma unroll
            for (int r = 0; r < 8; ++r) {
                int m = m0 + i * 16 + half * 8 + r;
                int n = nb + wn * 64 + j * 16 + l16;
                size_t idx = (size_t)m * N + n;
                float val = acc[i][j][r];
                if (EPI == 0) {
                    outf[idx] = val;
                } else if (EPI == 1) {
                    outf[idx] = 1.0f / (1.0f + __expf(-val));
                } else if (EPI == 2) {
                    outf[idx] = res[idx] + val;
                } else if (EPI == 3) {
                    outf[idx] = res[idx] + gate[idx] * val;
                } else {  // EPI == 4: relu^2 -> bf16
                    float t = fmaxf(val, 0.f);
                    outb[idx] = (bf16)(t * t);
                }
            }
        }
    }
}

// ---------------------------------------------------------------------------
// Host-side orchestration.
// ---------------------------------------------------------------------------
extern "C" void kernel_launch(void* const* d_in, const int* in_sizes, int n_in,
                              void* d_out, int out_size, void* d_ws, size_t ws_size,
                              hipStream_t stream) {
    const float* x          = (const float*)d_in[0];
    const float* ln1_g      = (const float*)d_in[1];
    const float* ln1_b      = (const float*)d_in[2];
    const float* ln2_g      = (const float*)d_in[3];
    const float* ln2_b      = (const float*)d_in[4];
    const float* tm_mix_k   = (const float*)d_in[5];
    const float* tm_mix_v   = (const float*)d_in[6];
    const float* tm_mix_r   = (const float*)d_in[7];
    const float* time_decay = (const float*)d_in[8];
    const float* time_first = (const float*)d_in[9];
    const float* wk         = (const float*)d_in[10];
    const float* wv         = (const float*)d_in[11];
    const float* wr         = (const float*)d_in[12];
    const float* wo         = (const float*)d_in[13];
    const float* cm_mix_k   = (const float*)d_in[14];
    const float* cm_mix_r   = (const float*)d_in[15];
    const float* cwk        = (const float*)d_in[16];
    const float* cwv        = (const float*)d_in[17];
    const float* cwr        = (const float*)d_in[18];
    float* out = (float*)d_out;

    // ---- Workspace layout (bytes) ----
    char* ws = (char*)d_ws;
    size_t o = 0;
    auto carve = [&](size_t bytes) { char* p = ws + o; o += bytes; return p; };
    const size_t DD2  = (size_t)Dn * Dn * sizeof(bf16);
    const size_t DF2  = (size_t)Dn * FFn * sizeof(bf16);
    const size_t MD2  = (size_t)Mn * Dn * sizeof(bf16);
    const size_t MD4  = (size_t)Mn * Dn * sizeof(float);

    bf16* wt_wk  = (bf16*)carve(DD2);
    bf16* wt_wv  = (bf16*)carve(DD2);
    bf16* wt_wr  = (bf16*)carve(DD2);
    bf16* wt_wo  = (bf16*)carve(DD2);
    bf16* wt_cwr = (bf16*)carve(DD2);
    bf16* wt_cwk = (bf16*)carve(DF2);   // [FF, D]
    bf16* wt_cwv = (bf16*)carve(DF2);   // [D, FF]
    float* h_buf = (float*)carve(MD4);  // LN output (reused for LN2)
    bf16* xk_buf = (bf16*)carve(MD2);   // later reused: rwkv bf16
    bf16* xv_buf = (bf16*)carve(MD2);   // later reused: xk2
    bf16* xr_buf = (bf16*)carve(MD2);   // later reused: xr2
    float* k_f   = (float*)carve(MD4);  // later reused: rr
    float* v_f   = (float*)carve(MD4);  // v_f + r_f contiguous -> later kk bf16
    float* r_f   = (float*)carve(MD4);
    float* x1    = (float*)carve(MD4);

    bf16* rwkv_b = xk_buf;
    bf16* xk2_b  = xv_buf;
    bf16* xr2_b  = xr_buf;
    float* rr_f  = k_f;
    bf16* kk_b   = (bf16*)v_f;          // [M, FF] bf16 = 128MB over v_f+r_f

    const int ewise_grid = (int)(((size_t)Mn * Dn + 255) / 256);

    // ---- 1. Weight convert + transpose (fp32 -> bf16, [K,N] -> [N,K]) ----
    {
        int gDD = (int)(((size_t)Dn * Dn + 255) / 256);
        int gDF = (int)(((size_t)Dn * FFn + 255) / 256);
        wconv_kernel<<<gDD, 256, 0, stream>>>(wk,  wt_wk,  Dn, Dn);
        wconv_kernel<<<gDD, 256, 0, stream>>>(wv,  wt_wv,  Dn, Dn);
        wconv_kernel<<<gDD, 256, 0, stream>>>(wr,  wt_wr,  Dn, Dn);
        wconv_kernel<<<gDD, 256, 0, stream>>>(wo,  wt_wo,  Dn, Dn);
        wconv_kernel<<<gDD, 256, 0, stream>>>(cwr, wt_cwr, Dn, Dn);
        wconv_kernel<<<gDF, 256, 0, stream>>>(cwk, wt_cwk, Dn, FFn);  // -> [FF,D]
        wconv_kernel<<<gDF, 256, 0, stream>>>(cwv, wt_cwv, FFn, Dn);  // -> [D,FF]
    }

    // ---- 2. LN1 + time-mix ----
    ln_kernel<<<Mn, 256, 0, stream>>>(x, ln1_g, ln1_b, h_buf);
    mix3_kernel<<<ewise_grid, 256, 0, stream>>>(h_buf, tm_mix_k, tm_mix_v, tm_mix_r,
                                                xk_buf, xv_buf, xr_buf);

    // ---- 3. k / v / r projections (bf16 WMMA) ----
    dim3 gD(Mn / BM, Dn / BN);     // (128, 8)
    wmma_gemm_kernel<0><<<gD, 256, 0, stream>>>(xk_buf, wt_wk, k_f, nullptr,
                                                nullptr, nullptr, Dn, Dn);
    wmma_gemm_kernel<0><<<gD, 256, 0, stream>>>(xv_buf, wt_wv, v_f, nullptr,
                                                nullptr, nullptr, Dn, Dn);
    wmma_gemm_kernel<1><<<gD, 256, 0, stream>>>(xr_buf, wt_wr, r_f, nullptr,
                                                nullptr, nullptr, Dn, Dn);

    // ---- 4. WKV recurrence (fuses r*wkv -> bf16) ----
    wkv_kernel<<<(Bn * Dn) / 256, 256, 0, stream>>>(k_f, v_f, r_f,
                                                    time_decay, time_first, rwkv_b);

    // ---- 5. Output projection + residual: x1 = x + (r*wkv) @ wo ----
    wmma_gemm_kernel<2><<<gD, 256, 0, stream>>>(rwkv_b, wt_wo, x1, nullptr,
                                                x, nullptr, Dn, Dn);

    // ---- 6. LN2 + channel-mix ----
    ln_kernel<<<Mn, 256, 0, stream>>>(x1, ln2_g, ln2_b, h_buf);
    mix2_kernel<<<ewise_grid, 256, 0, stream>>>(h_buf, cm_mix_k, cm_mix_r,
                                                xk2_b, xr2_b);

    // ---- 7. kk = relu(xk2 @ cwk)^2 (bf16), rr = sigmoid(xr2 @ cwr) ----
    dim3 gF(Mn / BM, FFn / BN);    // (128, 32)
    wmma_gemm_kernel<4><<<gF, 256, 0, stream>>>(xk2_b, wt_cwk, nullptr, kk_b,
                                                nullptr, nullptr, FFn, Dn);
    wmma_gemm_kernel<1><<<gD, 256, 0, stream>>>(xr2_b, wt_cwr, rr_f, nullptr,
                                                nullptr, nullptr, Dn, Dn);

    // ---- 8. out = x1 + rr * (kk @ cwv) ----
    wmma_gemm_kernel<3><<<gD, 256, 0, stream>>>(kk_b, wt_cwv, out, nullptr,
                                                x1, rr_f, Dn, FFn);

    (void)in_sizes; (void)n_in; (void)out_size; (void)ws_size;
}
